// _LoRA_qkv_17901423690471
// MI455X (gfx1250) — compile-verified, matrix operand order checked by reading, not run on hardware
//
#include <hip/hip_runtime.h>
#include <hip/hip_bf16.h>
#include <stdint.h>

// ---------------------------------------------------------------------------
// LoRA-fused QKV projection for MI455X (gfx1250, wave32, WMMA).
//   1) Fold the 4 weighted rank-16 LoRA branches into W_qkv -> bf16 W_eff (ws).
//   2) Convert x -> bf16 once (ws).  (HW lacks exposed cvt_pk_bf16_f32; doing
//      the 6-VALU-op/elem conversion per N-tile sweep would cost ~50us of
//      VALU issue vs the 22us memory budget, so convert exactly once.)
//   3) bf16 WMMA GEMM: out[32768,3072] = x_bf16 @ W_eff^T + b, with BOTH
//      tiles staged via global_load_async_to_lds_b128 (ASYNCcnt path) ->
//      hot loop is async-copy + ds_load + 8x v_wmma per K-step, minimal VALU.
// NT stores for the 384 MB output keep x_bf16 (64 MB) L2-resident across the
// 24 N-tile re-reads.  Double-buffered LDS, 1 barrier / K-step.
// ---------------------------------------------------------------------------

typedef __attribute__((ext_vector_type(16))) __bf16 v16bf;
typedef __attribute__((ext_vector_type(8)))  float  v8f;

union Frag16 {
    uint4 u[2];
    v16bf v;
};

__device__ __forceinline__ unsigned short f32_to_bf16_rne(float f) {
    unsigned int u = __builtin_bit_cast(unsigned int, f);
    u += 0x7FFFu + ((u >> 16) & 1u);
    return (unsigned short)(u >> 16);
}

__device__ __forceinline__ unsigned int f32x2_to_bf16x2(float lo, float hi) {
    return (unsigned int)f32_to_bf16_rne(lo) |
           ((unsigned int)f32_to_bf16_rne(hi) << 16);
}

// ---------------------------------------------------------------------------
// Kernel 1: W_eff[o,c] = W_qkv[o,c] (+ folded LoRA for q rows / v rows), bf16.
//   A_*: [4,16,1024], B_*: [4,1024,16]; branch weights {0.1,0.2,0.4,0.3}.
// ---------------------------------------------------------------------------
__global__ __launch_bounds__(256) void build_weff(
    const float* __restrict__ Wqkv,
    const float* __restrict__ Aq, const float* __restrict__ Bq,
    const float* __restrict__ Av, const float* __restrict__ Bv,
    unsigned short* __restrict__ Weff)
{
    const int idx = blockIdx.x * 256 + threadIdx.x;   // < 3072*1024
    const int o = idx >> 10;
    const int c = idx & 1023;

    float acc = Wqkv[idx];

    const float* Am = nullptr;
    const float* Bm = nullptr;
    int oo = 0;
    if (o < 1024)        { Am = Aq; Bm = Bq; oo = o; }
    else if (o >= 2048)  { Am = Av; Bm = Bv; oo = o - 2048; }

    if (Am) {
        const float wn[4] = {0.1f, 0.2f, 0.4f, 0.3f};
        #pragma unroll
        for (int n = 0; n < 4; ++n) {
            const float* Brow = Bm + ((size_t)n * 1024 + oo) * 16; // B[n,oo,r]
            const float* Acol = Am + (size_t)n * 16 * 1024 + c;    // A[n,r,c]
            float s = 0.f;
            #pragma unroll
            for (int r = 0; r < 16; ++r)
                s += Brow[r] * Acol[(size_t)r * 1024];
            acc += wn[n] * s;
        }
    }
    Weff[idx] = f32_to_bf16_rne(acc);
}

// ---------------------------------------------------------------------------
// Kernel 2: one-shot x fp32 -> bf16 (4 elems / thread).  NT loads so the
// soon-dead fp32 copy of x doesn't flush L2 right before the GEMM.
// ---------------------------------------------------------------------------
__global__ __launch_bounds__(256) void convert_x(
    const float* __restrict__ X, unsigned short* __restrict__ Xbf)
{
    const size_t i = ((size_t)blockIdx.x * 256 + threadIdx.x) * 4;
    const float4* p = (const float4*)(X + i);
    float4 f;
    f.x = __builtin_nontemporal_load(&((const float*)p)[0]);
    f.y = __builtin_nontemporal_load(&((const float*)p)[1]);
    f.z = __builtin_nontemporal_load(&((const float*)p)[2]);
    f.w = __builtin_nontemporal_load(&((const float*)p)[3]);
    union { unsigned int w[2]; uint2 u; } pk;
    pk.w[0] = f32x2_to_bf16x2(f.x, f.y);
    pk.w[1] = f32x2_to_bf16x2(f.z, f.w);
    *(uint2*)(Xbf + i) = pk.u;
}

// ---------------------------------------------------------------------------
// Kernel 3: out = x_bf16 @ W_eff^T + b, bf16 WMMA, double-buffered async LDS.
// Block tile 128x128, K-step 32; 8 waves/block, each wave owns 64x32
// (4x2 grid of 16x16 tiles -> 8x v_wmma_f32_16x16x32_bf16 per K-step).
// ---------------------------------------------------------------------------
#define BM     128
#define BN     128
#define BK     32
#define KSTEPS 32    // 1024 / BK
#define LDL    40    // padded LDS row stride in bf16 elements (80 B)

// async: global (64-bit vaddr) -> LDS (32-bit lds offset), 32 bytes.
// INST_OFFSET is added to BOTH global and LDS addresses, so one base pair
// covers both 16-byte chunks.  Tracked by ASYNCcnt.
__device__ __forceinline__ void async_copy_b256(
    unsigned int lds_off, const void* gaddr)
{
    asm volatile(
        "global_load_async_to_lds_b128 %0, %1, off\n\t"
        "global_load_async_to_lds_b128 %0, %1, off offset:16"
        :
        : "v"(lds_off), "v"((unsigned long long)(uintptr_t)gaddr)
        : "memory");
}

__device__ __forceinline__ void wait_async_all() {
    asm volatile("s_wait_asynccnt 0x0" ::: "memory");
}

__global__ __launch_bounds__(256) void gemm_wmma(
    const unsigned short* __restrict__ Xbf,  // [32768,1024] bf16
    const unsigned short* __restrict__ Weff, // [3072,1024] bf16
    const float* __restrict__ bias,          // [3072]
    float* __restrict__ Out)                 // [32768,3072]
{
    __shared__ unsigned short lsA[2][BM * LDL];
    __shared__ unsigned short lsB[2][BN * LDL];

    const int tid  = threadIdx.x;
    const int m0   = blockIdx.y * BM;
    const int n0   = blockIdx.x * BN;

    const int wave    = tid >> 5;
    const int lane    = tid & 31;
    const int laneLow = lane & 15;
    const int hiSel   = lane >> 4;           // 0 or 1

    const int wm = (wave & 1) * 64;          // wave tile M offset within block
    const int wn = (wave >> 1) * 32;         // wave tile N offset within block

    // cooperative stage-loader coords: 2 threads per row, 16 elems each
    const int rt   = tid >> 1;               // row 0..127
    const int half = (tid & 1) << 4;         // k-offset 0 or 16

    // per-thread staging addresses
    const unsigned short* srcA = Xbf  + (size_t)(m0 + rt) * 1024 + half;
    const unsigned short* srcB = Weff + (size_t)(n0 + rt) * 1024 + half;
    unsigned int ldsA[2], ldsB[2];
    #pragma unroll
    for (int b = 0; b < 2; ++b) {
        ldsA[b] = (unsigned int)(uintptr_t)(const void*)&lsA[b][rt * LDL + half];
        ldsB[b] = (unsigned int)(uintptr_t)(const void*)&lsB[b][rt * LDL + half];
    }

    v8f acc[4][2];
    #pragma unroll
    for (int i = 0; i < 4; ++i)
        #pragma unroll
        for (int j = 0; j < 2; ++j)
            acc[i][j] = v8f{};

    // ---- prologue: stage 0 (both tiles async) ----
    async_copy_b256(ldsA[0], srcA);
    async_copy_b256(ldsB[0], srcB);
    wait_async_all();
    __syncthreads();

    #pragma unroll 1
    for (int ks = 0; ks < KSTEPS; ++ks) {
        const int cur = ks & 1;
        const bool more = (ks + 1 < KSTEPS);

        // issue next-stage async copies into the other buffer
        // (free since iter ks-1's barrier); latency hidden by the WMMAs
        if (more) {
            const int k1 = (ks + 1) * BK;
            async_copy_b256(ldsA[cur ^ 1], srcA + k1);
            async_copy_b256(ldsB[cur ^ 1], srcB + k1);
        }

        // ---- fragments: 16 bf16 per lane per 16x16x32 operand tile ----
        Frag16 af[4], bf[2];
        #pragma unroll
        for (int i = 0; i < 4; ++i) {
            const unsigned short* p =
                &lsA[cur][(wm + i * 16 + laneLow) * LDL + hiSel * 16];
            af[i].u[0] = ((const uint4*)p)[0];
            af[i].u[1] = ((const uint4*)p)[1];
        }
        #pragma unroll
        for (int j = 0; j < 2; ++j) {
            const unsigned short* p =
                &lsB[cur][(wn + j * 16 + laneLow) * LDL + hiSel * 16];
            bf[j].u[0] = ((const uint4*)p)[0];
            bf[j].u[1] = ((const uint4*)p)[1];
        }

        // ---- 8 WMMAs per K-step ----
        #pragma unroll
        for (int i = 0; i < 4; ++i)
            #pragma unroll
            for (int j = 0; j < 2; ++j)
                acc[i][j] = __builtin_amdgcn_wmma_f32_16x16x32_bf16(
                    /*neg_a=*/false, af[i].v,
                    /*neg_b=*/false, bf[j].v,
                    /*c_mod=*/(short)0, acc[i][j],
                    /*reuse_a=*/false, /*reuse_b=*/false);

        // ---- wait for next stage, one barrier / iter ----
        if (more) {
            wait_async_all();
            __syncthreads();
        }
    }

    // ---- epilogue: + bias, non-temporal fp32 stores (keep x_bf16 in L2) ----
    // C/D layout: VGPR e -> M = e + 8*hiSel, N = laneLow (within 16x16 tile)
    #pragma unroll
    for (int j = 0; j < 2; ++j) {
        const int n  = n0 + wn + j * 16 + laneLow;
        const float bv = bias[n];
        #pragma unroll
        for (int i = 0; i < 4; ++i) {
            const int mb = m0 + wm + i * 16 + hiSel * 8;
            #pragma unroll
            for (int e = 0; e < 8; ++e) {
                __builtin_nontemporal_store(
                    acc[i][j][e] + bv,
                    &Out[(size_t)(mb + e) * 3072 + n]);
            }
        }
    }
}

// ---------------------------------------------------------------------------
extern "C" void kernel_launch(void* const* d_in, const int* in_sizes, int n_in,
                              void* d_out, int out_size, void* d_ws, size_t ws_size,
                              hipStream_t stream) {
    const float* x    = (const float*)d_in[0];  // [8,64,64,1024]
    const float* Wqkv = (const float*)d_in[1];  // [3072,1024]
    const float* bqkv = (const float*)d_in[2];  // [3072]
    const float* Aq   = (const float*)d_in[3];  // [4,16,1024]
    const float* Bq   = (const float*)d_in[4];  // [4,1024,16]
    const float* Av   = (const float*)d_in[5];  // [4,16,1024]
    const float* Bv   = (const float*)d_in[6];  // [4,1024,16]
    float* out = (float*)d_out;                 // [32768,3072]

    // workspace layout: [W_eff bf16: 6 MB][x bf16: 64 MB]
    unsigned short* Weff = (unsigned short*)d_ws;
    unsigned short* Xbf  = Weff + (size_t)3072 * 1024;

    build_weff<<<(3072 * 1024) / 256, 256, 0, stream>>>(Wqkv, Aq, Bq, Av, Bv, Weff);
    convert_x<<<(32768 * 1024) / (256 * 4), 256, 0, stream>>>(x, Xbf);

    dim3 grid(3072 / BN, 32768 / BM);  // (24, 256)
    gemm_wmma<<<grid, 256, 0, stream>>>(Xbf, Weff, bqkv, out);
}
